// SelfAttentionWOQ_90305982366358
// MI455X (gfx1250) — matrix operand order repaired and checked by reading, not compiled, not run on hardware
//
#include <hip/hip_runtime.h>
#include <hip/hip_bf16.h>

typedef __attribute__((ext_vector_type(16))) _Float16 v16h;
typedef __attribute__((ext_vector_type(8)))  _Float16 v8h;
typedef __attribute__((ext_vector_type(8)))  float    v8f;
typedef __attribute__((ext_vector_type(4)))  int      v4i;

union FragH { v16h v; v8h h[2]; _Float16 e[16]; };

#define WMMA(a, b, c) \
  __builtin_amdgcn_wmma_f32_16x16x32_f16(false, (a), false, (b), (short)0, (c), false, false)

#if __has_builtin(__builtin_amdgcn_global_load_async_to_lds_b128)
#define ASYNC_LDS 1
typedef __attribute__((address_space(1))) v4i* gv4i_ptr;
typedef __attribute__((address_space(3))) v4i* lv4i_ptr;
#endif

__device__ __forceinline__ void async_copy_b128(const _Float16* g, _Float16* l) {
#ifdef ASYNC_LDS
  __builtin_amdgcn_global_load_async_to_lds_b128(
      (gv4i_ptr)g, (lv4i_ptr)l, 0, 0);
#else
  *(v8h*)l = *(const v8h*)g;
#endif
}

__device__ __forceinline__ void async_wait_all() {
#ifdef ASYNC_LDS
#if __has_builtin(__builtin_amdgcn_s_wait_asynccnt)
  __builtin_amdgcn_s_wait_asynccnt(0);
#else
  asm volatile("s_wait_asynccnt 0x0" ::: "memory");
#endif
#endif
}

// ---------------------------------------------------------------------------
// Convert fp32 activations to f16.
// ---------------------------------------------------------------------------
__global__ void f32_to_f16_kernel(const float* __restrict__ in,
                                  _Float16* __restrict__ out, int n) {
  int i = blockIdx.x * blockDim.x + threadIdx.x;
  int stride = gridDim.x * blockDim.x;
  for (; i < n; i += stride) out[i] = (_Float16)in[i];
}

// ---------------------------------------------------------------------------
// Per-output-channel symmetric int8 fake quant; store quantized integer
// values (exact in f16) packed directly in the WMMA B-fragment layout:
// tile t = (k/32)*(N/16) + n/16 ; lane = ((k>>4)&1)*16 + (n&15) ; j = k&15
// element offset = t*512 + lane*16 + j  (so a lane's 16 halves are contiguous)
// ---------------------------------------------------------------------------
__global__ void quant_pack_kernel(const float* __restrict__ W,  // N x K (out,in)
                                  _Float16* __restrict__ Bp,
                                  float* __restrict__ scales, int N, int K) {
  const int n = blockIdx.x;
  const int tid = threadIdx.x;
  __shared__ float red[256];
  const float* row = W + (size_t)n * K;
  float m = 0.f;
  for (int k = tid; k < K; k += 256) m = fmaxf(m, fabsf(row[k]));
  red[tid] = m;
  __syncthreads();
  for (int s = 128; s > 0; s >>= 1) {
    if (tid < s) red[tid] = fmaxf(red[tid], red[tid + s]);
    __syncthreads();
  }
  const float EPS = 1.1920928955078125e-07f;
  float scale = fmaxf(red[0] / 127.f, EPS);
  if (tid == 0) scales[n] = scale;
  float inv = 1.f / scale;
  const int ntiles = N >> 4;
  for (int k = tid; k < K; k += 256) {
    float q = rintf(row[k] * inv);
    q = fminf(127.f, fmaxf(-127.f, q));
    int t = (k >> 5) * ntiles + (n >> 4);
    int lane = ((k >> 4) & 1) * 16 + (n & 15);
    int j = k & 15;
    Bp[(size_t)t * 512 + lane * 16 + j] = (_Float16)q;
  }
}

// ---------------------------------------------------------------------------
// GEMM: out[m][n] = (sum_k A[m][k]*wq_int[n][k]) * scale[n] + bias[n]
// A: M x K row-major f16. Bp: packed B fragments. Each wave -> 16x64 tile.
// ---------------------------------------------------------------------------
template <typename OutT>
__global__ void gemm_wq_kernel(const _Float16* __restrict__ A,
                               const _Float16* __restrict__ Bp,
                               const float* __restrict__ scale,
                               const float* __restrict__ bias,
                               OutT* __restrict__ out, int M, int N, int K) {
  const int lane = threadIdx.x & 31;
  const int halfw = lane >> 4;
  const int l16 = lane & 15;
  const int wid = (blockIdx.x * blockDim.x + threadIdx.x) >> 5;
  const int ncol4 = N >> 6;
  const int m0 = (wid / ncol4) << 4;
  const int nb = (wid % ncol4) << 6;
  if (m0 >= M) return;
  const int ntiles = N >> 4;

  v8f acc[4] = {};
  const _Float16* arow = A + (size_t)(m0 + l16) * K + halfw * 16;

  for (int k0 = 0; k0 < K; k0 += 32) {
    FragH a;
    a.h[0] = *(const v8h*)(arow + k0);
    a.h[1] = *(const v8h*)(arow + k0 + 8);
    __builtin_prefetch(arow + k0 + 64, 0, 0);
    const _Float16* bt =
        Bp + ((size_t)(k0 >> 5) * ntiles + (nb >> 4)) * 512 + lane * 16;
#pragma unroll
    for (int t = 0; t < 4; ++t) {
      FragH b;
      b.h[0] = *(const v8h*)(bt + t * 512);
      b.h[1] = *(const v8h*)(bt + t * 512 + 8);
      acc[t] = WMMA(a.v, b.v, acc[t]);
    }
  }

#pragma unroll
  for (int t = 0; t < 4; ++t) {
    int n = nb + t * 16 + l16;
    float s = scale[n];
    float bb = bias[n];
#pragma unroll
    for (int r = 0; r < 8; ++r) {
      int m = m0 + r + 8 * halfw;
      out[(size_t)m * N + n] = (OutT)(acc[t][r] * s + bb);
    }
  }
}

// ---------------------------------------------------------------------------
// Flash attention: B=2, S=2048, H=16, D=64. Q/K/V/O stored as (B*S, 1024)
// f16 with head h occupying columns [h*64, h*64+64).
// Block = 128 threads (4 waves); each wave owns 16 query rows; block covers
// 64 query rows of one (b,h). Keys processed in chunks of 64:
//   K chunk async-copied to LDS (ASYNCcnt path), V transposed into LDS.
// Softmax scale (1/8, exact pow2) is folded into the Q fragments.
// ---------------------------------------------------------------------------
__global__ void attention_kernel(const _Float16* __restrict__ Q,
                                 const _Float16* __restrict__ Kg,
                                 const _Float16* __restrict__ Vg,
                                 _Float16* __restrict__ Og) {
  const int S = 2048, C = 1024, Dh = 64;
  const int sblk = blockIdx.x;  // S/64 = 32
  const int h = blockIdx.y;     // 16
  const int b = blockIdx.z;     // 2
  const int tid = threadIdx.x;
  const int wave = tid >> 5;
  const int lane = tid & 31;
  const int halfw = lane >> 4;
  const int l16 = lane & 15;

  __shared__ __align__(16) _Float16 Kl[64][64];      // keys row-major (t, d)
  __shared__ __align__(16) _Float16 Vt[64][64];      // V transposed   (d, t)
  __shared__ __align__(16) _Float16 Pl[4][16][64];   // per-wave P staging

  const size_t rowbase = (size_t)b * S;
  const int s0 = sblk * 64 + wave * 16;

  // Q A-fragments for d-chunks [0,32) and [32,64); pre-scaled by 1/sqrt(D).
  FragH qf0, qf1;
  {
    const _Float16* qrow = Q + (rowbase + s0 + l16) * C + h * Dh + halfw * 16;
    qf0.h[0] = *(const v8h*)(qrow);
    qf0.h[1] = *(const v8h*)(qrow + 8);
    qf1.h[0] = *(const v8h*)(qrow + 32);
    qf1.h[1] = *(const v8h*)(qrow + 40);
#pragma unroll
    for (int j = 0; j < 16; ++j) {
      qf0.e[j] = qf0.e[j] * (_Float16)0.125f;  // exact power-of-two scale
      qf1.e[j] = qf1.e[j] * (_Float16)0.125f;
    }
  }

  v8f o[4] = {};
  float mrow[8], lrow[8];
#pragma unroll
  for (int r = 0; r < 8; ++r) { mrow[r] = -1e30f; lrow[r] = 0.f; }

  for (int t0 = 0; t0 < S; t0 += 64) {
    // Cooperative stage: each thread owns half a key row (32 halves).
    {
      int row = tid >> 1;             // key 0..63
      int colbase = (tid & 1) << 5;   // d 0 or 32
      const _Float16* kp = Kg + (rowbase + t0 + row) * C + h * Dh + colbase;
#pragma unroll
      for (int j = 0; j < 4; ++j)
        async_copy_b128(kp + j * 8, &Kl[row][colbase + j * 8]);
      const _Float16* vp = Vg + (rowbase + t0 + row) * C + h * Dh + colbase;
      FragH vv;
      vv.h[0] = *(const v8h*)vp;
      vv.h[1] = *(const v8h*)(vp + 8);
      FragH vw;
      vw.h[0] = *(const v8h*)(vp + 16);
      vw.h[1] = *(const v8h*)(vp + 24);
#pragma unroll
      for (int j = 0; j < 16; ++j) Vt[colbase + j][row] = vv.e[j];
#pragma unroll
      for (int j = 0; j < 16; ++j) Vt[colbase + 16 + j][row] = vw.e[j];
    }
    async_wait_all();
    __syncthreads();

    // S = Q K^T : four 16x16 score tiles covering keys t0 .. t0+63
    v8f sc[4] = {};
#pragma unroll
    for (int nt = 0; nt < 4; ++nt) {
      FragH kb;
      kb.h[0] = *(const v8h*)&Kl[nt * 16 + l16][halfw * 16];
      kb.h[1] = *(const v8h*)&Kl[nt * 16 + l16][halfw * 16 + 8];
      sc[nt] = WMMA(qf0.v, kb.v, sc[nt]);
      kb.h[0] = *(const v8h*)&Kl[nt * 16 + l16][32 + halfw * 16];
      kb.h[1] = *(const v8h*)&Kl[nt * 16 + l16][32 + halfw * 16 + 8];
      sc[nt] = WMMA(qf1.v, kb.v, sc[nt]);
    }

    // Online softmax. Row m = r + 8*halfw lives across the 16 lanes of this
    // half-wave; xor masks 1..8 stay inside the half (wave32).
#pragma unroll
    for (int r = 0; r < 8; ++r) {
      float x0 = sc[0][r], x1 = sc[1][r], x2 = sc[2][r], x3 = sc[3][r];
      float mx = fmaxf(fmaxf(x0, x1), fmaxf(x2, x3));
#pragma unroll
      for (int d = 1; d < 16; d <<= 1) mx = fmaxf(mx, __shfl_xor(mx, d));
      float mnew = fmaxf(mrow[r], mx);
      float alpha = __expf(mrow[r] - mnew);
      float p0 = __expf(x0 - mnew);
      float p1 = __expf(x1 - mnew);
      float p2 = __expf(x2 - mnew);
      float p3 = __expf(x3 - mnew);
      float ps = (p0 + p1) + (p2 + p3);
#pragma unroll
      for (int d = 1; d < 16; d <<= 1) ps += __shfl_xor(ps, d);
      lrow[r] = lrow[r] * alpha + ps;
      mrow[r] = mnew;
      o[0][r] *= alpha;
      o[1][r] *= alpha;
      o[2][r] *= alpha;
      o[3][r] *= alpha;
      Pl[wave][r + 8 * halfw][l16] = (_Float16)p0;
      Pl[wave][r + 8 * halfw][16 + l16] = (_Float16)p1;
      Pl[wave][r + 8 * halfw][32 + l16] = (_Float16)p2;
      Pl[wave][r + 8 * halfw][48 + l16] = (_Float16)p3;
    }

    // O += P V : P reloaded in A layout (two K=32 chunks), V B-frags from
    // transposed LDS (contiguous 32B per lane).
    FragH pa0, pa1;
    pa0.h[0] = *(const v8h*)&Pl[wave][l16][halfw * 16];
    pa0.h[1] = *(const v8h*)&Pl[wave][l16][halfw * 16 + 8];
    pa1.h[0] = *(const v8h*)&Pl[wave][l16][32 + halfw * 16];
    pa1.h[1] = *(const v8h*)&Pl[wave][l16][32 + halfw * 16 + 8];
#pragma unroll
    for (int t = 0; t < 4; ++t) {
      FragH vb;
      vb.h[0] = *(const v8h*)&Vt[t * 16 + l16][halfw * 16];
      vb.h[1] = *(const v8h*)&Vt[t * 16 + l16][halfw * 16 + 8];
      o[t] = WMMA(pa0.v, vb.v, o[t]);
      vb.h[0] = *(const v8h*)&Vt[t * 16 + l16][32 + halfw * 16];
      vb.h[1] = *(const v8h*)&Vt[t * 16 + l16][32 + halfw * 16 + 8];
      o[t] = WMMA(pa1.v, vb.v, o[t]);
    }
    __syncthreads();
  }

  // Normalize and store O (f16, (B,S,H,D) layout).
#pragma unroll
  for (int r = 0; r < 8; ++r) {
    float invl = 1.f / lrow[r];
    int srow = s0 + r + 8 * halfw;
    _Float16* op = Og + (rowbase + srow) * C + h * Dh;
    op[l16] = (_Float16)(o[0][r] * invl);
    op[16 + l16] = (_Float16)(o[1][r] * invl);
    op[32 + l16] = (_Float16)(o[2][r] * invl);
    op[48 + l16] = (_Float16)(o[3][r] * invl);
  }
}

// ---------------------------------------------------------------------------
extern "C" void kernel_launch(void* const* d_in, const int* in_sizes, int n_in,
                              void* d_out, int out_size, void* d_ws,
                              size_t ws_size, hipStream_t stream) {
  const float* x  = (const float*)d_in[0];
  const float* wq = (const float*)d_in[1];
  const float* bq = (const float*)d_in[2];
  const float* wk = (const float*)d_in[3];
  const float* bk = (const float*)d_in[4];
  const float* wv = (const float*)d_in[5];
  const float* bv = (const float*)d_in[6];
  const float* wp = (const float*)d_in[7];
  const float* bp = (const float*)d_in[8];
  float* out = (float*)d_out;

  const int M = 4096, N = 1024, K = 1024;  // B*S=4096, C=1024

  char* ws = (char*)d_ws;
  size_t off = 0;
  auto alloc = [&](size_t bytes) -> char* {
    char* p = ws + off;
    off += (bytes + 255) & ~(size_t)255;
    return p;
  };
  _Float16* Xh  = (_Float16*)alloc((size_t)M * K * 2);
  _Float16* Wqp = (_Float16*)alloc((size_t)K * N * 2);
  _Float16* Wkp = (_Float16*)alloc((size_t)K * N * 2);
  _Float16* Wvp = (_Float16*)alloc((size_t)K * N * 2);
  _Float16* Wpp = (_Float16*)alloc((size_t)K * N * 2);
  float*    scl = (float*)alloc(4 * 1024 * sizeof(float));
  _Float16* Qh  = (_Float16*)alloc((size_t)M * N * 2);
  _Float16* Kh  = (_Float16*)alloc((size_t)M * N * 2);
  _Float16* Vh  = (_Float16*)alloc((size_t)M * N * 2);
  _Float16* Oh  = (_Float16*)alloc((size_t)M * N * 2);

  f32_to_f16_kernel<<<4096, 256, 0, stream>>>(x, Xh, M * K);

  quant_pack_kernel<<<1024, 256, 0, stream>>>(wq, Wqp, scl + 0,    N, K);
  quant_pack_kernel<<<1024, 256, 0, stream>>>(wk, Wkp, scl + 1024, N, K);
  quant_pack_kernel<<<1024, 256, 0, stream>>>(wv, Wvp, scl + 2048, N, K);
  quant_pack_kernel<<<1024, 256, 0, stream>>>(wp, Wpp, scl + 3072, N, K);

  // waves = (M/16)*(N/64) = 4096 ; 8 waves per 256-thread block -> 512 blocks
  const int gemm_blocks = (M / 16) * (N / 64) / 8;
  gemm_wq_kernel<_Float16><<<gemm_blocks, 256, 0, stream>>>(
      Xh, Wqp, scl + 0, bq, Qh, M, N, K);
  gemm_wq_kernel<_Float16><<<gemm_blocks, 256, 0, stream>>>(
      Xh, Wkp, scl + 1024, bk, Kh, M, N, K);
  gemm_wq_kernel<_Float16><<<gemm_blocks, 256, 0, stream>>>(
      Xh, Wvp, scl + 2048, bv, Vh, M, N, K);

  attention_kernel<<<dim3(32, 16, 2), 128, 0, stream>>>(Qh, Kh, Vh, Oh);

  gemm_wq_kernel<float><<<gemm_blocks, 256, 0, stream>>>(
      Oh, Wpp, scl + 3072, bp, out, M, N, K);
}
